// T5SelfAttention_77927886619149
// MI455X (gfx1250) — compile-verified, matrix operand order checked by reading, not run on hardware
//
#include <hip/hip_runtime.h>
#include <hip/hip_bf16.h>

// ---------------- problem constants ----------------
#define BATCH   4
#define SEQ     1024
#define DMODEL  1024
#define HEADS   16
#define HDIM    64
#define MROWS   (BATCH * SEQ)          // 4096
#define NBUCKETS 32

// ---------------- types ----------------
typedef __bf16 bf16_t;
typedef bf16_t bf16x16 __attribute__((ext_vector_type(16)));
typedef float  f32x8   __attribute__((ext_vector_type(8)));

// float -> bf16 with round-to-nearest-even (explicit, no reliance on cast lowering)
static __device__ __forceinline__ bf16_t f2bf(float f) {
  unsigned u = __builtin_bit_cast(unsigned, f);
  u += 0x7FFFu + ((u >> 16) & 1u);
  unsigned short h = (unsigned short)(u >> 16);
  return __builtin_bit_cast(bf16_t, h);
}

static __device__ __forceinline__ f32x8 wmma_bf16(bf16x16 a, bf16x16 b, f32x8 c) {
  // D(16x16,f32) = A(16x32,bf16) x B(32x16,bf16) + C
  return __builtin_amdgcn_wmma_f32_16x16x32_bf16(false, a, false, b, (short)0, c,
                                                 false, false);
}

// Load a 16x32 bf16 fragment (A layout; B uses the same layout fed with B^T rows).
// Per ISA 7.12.2: lanes 0-15 row M=lane hold K = 0..7 then 16..23;
// lanes 16-31 row M=lane-16 hold K = 8..15 then 24..31.
static __device__ __forceinline__ bf16x16 load_frag(const bf16_t* tile, int stride) {
  const int lane = (int)(threadIdx.x & 31u);
  const int r    = lane & 15;
  const int kb   = (lane >> 4) << 3;
  const bf16_t* p = tile + r * stride + kb;
  union { uint4 u[2]; bf16x16 v; } f;
  f.u[0] = *reinterpret_cast<const uint4*>(p);
  f.u[1] = *reinterpret_cast<const uint4*>(p + 16);
  return f.v;
}

// ---------------- stage 1a: X (f32) -> bf16 ----------------
__global__ void __launch_bounds__(256) k_convert_x(const float* __restrict__ X,
                                                   bf16_t* __restrict__ Xb) {
  int i = (int)(blockIdx.x * blockDim.x + threadIdx.x) * 4;
  float4 v = *reinterpret_cast<const float4*>(X + i);
  Xb[i + 0] = f2bf(v.x);
  Xb[i + 1] = f2bf(v.y);
  Xb[i + 2] = f2bf(v.z);
  Xb[i + 3] = f2bf(v.w);
}

// ---------------- stage 1b: transpose + convert weights (LDS tile) ----------------
// Wt[n][k] = W[k][n], output bf16. grid.z selects which weight.
__global__ void __launch_bounds__(256) k_transpose_w(const float* __restrict__ w0,
                                                     const float* __restrict__ w1,
                                                     const float* __restrict__ w2,
                                                     const float* __restrict__ w3,
                                                     bf16_t* __restrict__ wt_base) {
  __shared__ float tile[32][33];
  const float* W = (blockIdx.z == 0) ? w0 : (blockIdx.z == 1) ? w1
                  : (blockIdx.z == 2) ? w2 : w3;
  bf16_t* Wt = wt_base + (size_t)blockIdx.z * DMODEL * DMODEL;

  int x  = blockIdx.x * 32 + threadIdx.x;   // n
  int y0 = blockIdx.y * 32 + threadIdx.y;   // k
#pragma unroll
  for (int i = 0; i < 32; i += 8)
    tile[threadIdx.y + i][threadIdx.x] = W[(size_t)(y0 + i) * DMODEL + x];
  __syncthreads();
  int xt  = blockIdx.y * 32 + threadIdx.x;  // k
  int yt0 = blockIdx.x * 32 + threadIdx.y;  // n
#pragma unroll
  for (int i = 0; i < 32; i += 8)
    Wt[(size_t)(yt0 + i) * DMODEL + xt] = f2bf(tile[threadIdx.x][threadIdx.y + i]);
}

// ---------------- stage 1c: T5 relative-position bias delta table ----------------
// bias_delta[h][d] for d = (k - q) + 1023, d in [0,2046] (slot 2047 duplicated).
__global__ void __launch_bounds__(256) k_build_bias(const float* __restrict__ rpb,
                                                    float* __restrict__ bias_delta) {
  int idx = (int)(blockIdx.x * blockDim.x + threadIdx.x);
  int h = idx >> 11;
  int d = idx & 2047;
  if (h >= HEADS) return;
  int delta = d - 1023;
  if (delta > 1023) delta = 1023;
  int bucket = (delta > 0) ? 16 : 0;
  int rp = delta < 0 ? -delta : delta;
  if (rp < 8) {
    bucket += rp;
  } else {
    float lg = __logf((float)rp * 0.125f) * (8.0f / __logf(16.0f)); // log(128/8)=log16
    int large = 8 + (int)lg;
    if (large > 15) large = 15;
    bucket += large;
  }
  bias_delta[idx] = rpb[bucket * HEADS + h];
}

// ---------------- stage 2: fused QKV WMMA GEMM ----------------
// out = X[4096x1024] * W; per workgroup: 512 rows x 64 cols; 8 waves, each 64x64
// (4x4 WMMA tiles -> 16 WMMA per 8 fragment loads).
// z = 0 -> Q [B,H,L,64]; z = 1 -> K [B,H,L,64]; z = 2 -> V^T [B,H,64,L]
__global__ void __launch_bounds__(256) k_gemm_qkv(const bf16_t* __restrict__ X,
                                                  const bf16_t* __restrict__ wqt,
                                                  const bf16_t* __restrict__ wkt,
                                                  const bf16_t* __restrict__ wvt,
                                                  bf16_t* __restrict__ qo,
                                                  bf16_t* __restrict__ ko,
                                                  bf16_t* __restrict__ vt) {
  const int K = DMODEL;
  const int z = (int)blockIdx.z;
  const bf16_t* Bt = (z == 0) ? wqt : (z == 1) ? wkt : wvt;

  const int wave = (int)(threadIdx.x >> 5);
  const int row0 = (int)blockIdx.x * 512 + wave * 64;
  const int col0 = (int)blockIdx.y * 64;

  const f32x8 z8 = {0.f, 0.f, 0.f, 0.f, 0.f, 0.f, 0.f, 0.f};
  f32x8 acc[4][4];
#pragma unroll
  for (int i = 0; i < 4; ++i)
#pragma unroll
    for (int j = 0; j < 4; ++j) acc[i][j] = z8;

  for (int k = 0; k < K; k += 32) {
    __builtin_prefetch(X + (size_t)row0 * K + k + 512, 0, 1);
    bf16x16 af[4];
#pragma unroll
    for (int i = 0; i < 4; ++i)
      af[i] = load_frag(X + (size_t)(row0 + 16 * i) * K + k, K);
#pragma unroll
    for (int j = 0; j < 4; ++j) {
      bf16x16 b = load_frag(Bt + (size_t)(col0 + 16 * j) * K + k, K);
#pragma unroll
      for (int i = 0; i < 4; ++i) acc[i][j] = wmma_bf16(af[i], b, acc[i][j]);
    }
  }

  const int lane = (int)(threadIdx.x & 31u);
  const int col  = lane & 15;
  const int rb   = (lane >> 4) << 3;
#pragma unroll
  for (int i = 0; i < 4; ++i) {
#pragma unroll
    for (int j = 0; j < 4; ++j) {
#pragma unroll
      for (int r = 0; r < 8; ++r) {
        int m = row0 + 16 * i + rb + r;
        int n = col0 + 16 * j + col;
        int b  = m >> 10, l = m & 1023;
        int h  = n >> 6,  hd = n & 63;
        bf16_t val = f2bf(acc[i][j][r]);
        if (z == 2) {
          vt[((size_t)(b * HEADS + h) * HDIM + hd) * SEQ + l] = val;
        } else {
          bf16_t* o = (z == 0) ? qo : ko;
          o[((size_t)(b * HEADS + h) * SEQ + l) * HDIM + hd] = val;
        }
      }
    }
  }
}

// ---------------- stage 3: flash attention with T5 bias ----------------
// grid: (SEQ/64, BATCH*HEADS); block: 128 (4 waves); wave handles 16 q-rows.
__global__ void __launch_bounds__(128) k_attention(const bf16_t* __restrict__ q,
                                                   const bf16_t* __restrict__ k,
                                                   const bf16_t* __restrict__ vt,
                                                   const float* __restrict__ bias_delta,
                                                   bf16_t* __restrict__ ctx) {
  __shared__ __align__(16) float sbias[2048];
  __shared__ __align__(16) bf16_t pbuf[4][16][32];

  const int bh = (int)blockIdx.y;
  const int b = bh >> 4, h = bh & 15;
  const int q0wg = (int)blockIdx.x * 64;

  // ---- stage bias row for this head into LDS via CDNA5 async copy ----
  {
    unsigned lds0 = (unsigned)(uintptr_t)(&sbias[0]);
    const float* gsrc = bias_delta + (size_t)h * 2048;
    const int t = (int)threadIdx.x;
#pragma unroll
    for (int it = 0; it < 4; ++it) {
      int e16 = (it * 128 + t) * 16;  // byte offset, 16B per lane per pass
      unsigned lda = lds0 + (unsigned)e16;
      unsigned long long ga =
          (unsigned long long)(uintptr_t)gsrc + (unsigned long long)e16;
      asm volatile("global_load_async_to_lds_b128 %0, %1, off"
                   :: "v"(lda), "v"(ga) : "memory");
    }
    asm volatile("s_wait_asynccnt 0" ::: "memory");
  }
  __syncthreads();

  const int wv   = (int)(threadIdx.x >> 5);
  const int lane = (int)(threadIdx.x & 31u);
  const int col  = lane & 15;
  const int rb   = (lane >> 4) << 3;
  const int q0   = q0wg + wv * 16;

  const bf16_t* qh = q  + ((size_t)bh * SEQ + q0) * HDIM;
  const bf16_t* kh = k  + (size_t)bh * SEQ * HDIM;
  const bf16_t* vh = vt + (size_t)bh * HDIM * SEQ;

  bf16x16 qf0 = load_frag(qh, HDIM);       // d = 0..31
  bf16x16 qf1 = load_frag(qh + 32, HDIM);  // d = 32..63

  const f32x8 z8 = {0.f, 0.f, 0.f, 0.f, 0.f, 0.f, 0.f, 0.f};
  f32x8 o[4];
#pragma unroll
  for (int j = 0; j < 4; ++j) o[j] = z8;
  float mrow[8], lrow[8];
#pragma unroll
  for (int r = 0; r < 8; ++r) { mrow[r] = -3.0e38f; lrow[r] = 0.0f; }

  for (int kp = 0; kp < SEQ; kp += 32) {
    // ---- scores S[16 x 32] = Q * K^T (scale = 1.0, T5 convention) ----
    f32x8 s0 = z8, s1 = z8;
    {
      bf16x16 kf;
      kf = load_frag(kh + (size_t)kp * HDIM, HDIM);             s0 = wmma_bf16(qf0, kf, s0);
      kf = load_frag(kh + (size_t)kp * HDIM + 32, HDIM);        s0 = wmma_bf16(qf1, kf, s0);
      kf = load_frag(kh + (size_t)(kp + 16) * HDIM, HDIM);      s1 = wmma_bf16(qf0, kf, s1);
      kf = load_frag(kh + (size_t)(kp + 16) * HDIM + 32, HDIM); s1 = wmma_bf16(qf1, kf, s1);
    }
    // ---- add relative position bias: delta = kpos - qrow ----
    const int base0 = kp + col + 1023 - q0 - rb;
#pragma unroll
    for (int r = 0; r < 8; ++r) {
      s0[r] += sbias[base0 - r];
      s1[r] += sbias[base0 - r + 16];
    }
    // ---- online softmax: row max over 32 cols (16 lanes per row group) ----
    float pm[8];
#pragma unroll
    for (int r = 0; r < 8; ++r) pm[r] = fmaxf(s0[r], s1[r]);
#pragma unroll
    for (int msk = 1; msk < 16; msk <<= 1)
#pragma unroll
      for (int r = 0; r < 8; ++r) pm[r] = fmaxf(pm[r], __shfl_xor(pm[r], msk, 32));
    float al[8];
#pragma unroll
    for (int r = 0; r < 8; ++r) {
      float mn = fmaxf(mrow[r], pm[r]);
      al[r] = __expf(mrow[r] - mn);
      mrow[r] = mn;
    }
    float rs[8];
#pragma unroll
    for (int r = 0; r < 8; ++r) {
      s0[r] = __expf(s0[r] - mrow[r]);
      s1[r] = __expf(s1[r] - mrow[r]);
      rs[r] = s0[r] + s1[r];
    }
#pragma unroll
    for (int msk = 1; msk < 16; msk <<= 1)
#pragma unroll
      for (int r = 0; r < 8; ++r) rs[r] += __shfl_xor(rs[r], msk, 32);
#pragma unroll
    for (int r = 0; r < 8; ++r) lrow[r] = lrow[r] * al[r] + rs[r];
#pragma unroll
    for (int j = 0; j < 4; ++j)
#pragma unroll
      for (int r = 0; r < 8; ++r) o[j][r] *= al[r];
    // ---- stage P through LDS: C-layout -> A-layout, f32 -> bf16 ----
#pragma unroll
    for (int r = 0; r < 8; ++r) {
      pbuf[wv][rb + r][col]      = f2bf(s0[r]);
      pbuf[wv][rb + r][col + 16] = f2bf(s1[r]);
    }
    asm volatile("s_wait_dscnt 0" ::: "memory");  // wave-local LDS RAW fence
    bf16x16 pf = load_frag(&pbuf[wv][0][0], 32);
    // ---- O += P * V  (V^T rows are contiguous) ----
#pragma unroll
    for (int j = 0; j < 4; ++j) {
      bf16x16 vf = load_frag(vh + (size_t)(16 * j) * SEQ + kp, SEQ);
      o[j] = wmma_bf16(pf, vf, o[j]);
    }
  }

  // ---- normalize and store ctx [4096 x 1024] (row = b*SEQ+l, col = h*64+d) ----
#pragma unroll
  for (int j = 0; j < 4; ++j)
#pragma unroll
    for (int r = 0; r < 8; ++r) {
      int qrow = q0 + rb + r;
      ctx[((size_t)(b * SEQ + qrow)) * (HEADS * HDIM) + h * HDIM + 16 * j + col] =
          f2bf(o[j][r] / lrow[r]);
    }
}

// ---------------- stage 4: output projection -> f32 ----------------
__global__ void __launch_bounds__(256) k_gemm_out(const bf16_t* __restrict__ A,
                                                  const bf16_t* __restrict__ Bt,
                                                  float* __restrict__ Out) {
  const int K = DMODEL, N = DMODEL;
  const int wave = (int)(threadIdx.x >> 5);
  const int row0 = (int)blockIdx.x * 512 + wave * 64;
  const int col0 = (int)blockIdx.y * 64;

  const f32x8 z8 = {0.f, 0.f, 0.f, 0.f, 0.f, 0.f, 0.f, 0.f};
  f32x8 acc[4][4];
#pragma unroll
  for (int i = 0; i < 4; ++i)
#pragma unroll
    for (int j = 0; j < 4; ++j) acc[i][j] = z8;

  for (int k = 0; k < K; k += 32) {
    __builtin_prefetch(A + (size_t)row0 * K + k + 512, 0, 1);
    bf16x16 af[4];
#pragma unroll
    for (int i = 0; i < 4; ++i)
      af[i] = load_frag(A + (size_t)(row0 + 16 * i) * K + k, K);
#pragma unroll
    for (int j = 0; j < 4; ++j) {
      bf16x16 bfr = load_frag(Bt + (size_t)(col0 + 16 * j) * K + k, K);
#pragma unroll
      for (int i = 0; i < 4; ++i) acc[i][j] = wmma_bf16(af[i], bfr, acc[i][j]);
    }
  }
  const int lane = (int)(threadIdx.x & 31u);
  const int col  = lane & 15;
  const int rb   = (lane >> 4) << 3;
#pragma unroll
  for (int i = 0; i < 4; ++i)
#pragma unroll
    for (int j = 0; j < 4; ++j)
#pragma unroll
      for (int r = 0; r < 8; ++r)
        Out[(size_t)(row0 + 16 * i + rb + r) * N + col0 + 16 * j + col] = acc[i][j][r];
}

// ---------------- host side ----------------
extern "C" void kernel_launch(void* const* d_in, const int* in_sizes, int n_in,
                              void* d_out, int out_size, void* d_ws, size_t ws_size,
                              hipStream_t stream) {
  (void)in_sizes; (void)n_in; (void)out_size; (void)ws_size;
  const float* hs  = (const float*)d_in[0];
  const float* wq  = (const float*)d_in[1];
  const float* wk  = (const float*)d_in[2];
  const float* wv  = (const float*)d_in[3];
  const float* wo  = (const float*)d_in[4];
  const float* rpb = (const float*)d_in[5];
  float* out = (float*)d_out;

  const size_t MB = 1024 * 1024;
  char* ws = (char*)d_ws;
  bf16_t* xb   = (bf16_t*)(ws + 0);        // 8 MB  X bf16 [4096,1024]
  bf16_t* wt   = (bf16_t*)(ws + 8 * MB);   // 8 MB  wq^T,wk^T,wv^T,wo^T bf16
  bf16_t* wqt  = wt + 0 * (size_t)DMODEL * DMODEL;
  bf16_t* wkt  = wt + 1 * (size_t)DMODEL * DMODEL;
  bf16_t* wvt  = wt + 2 * (size_t)DMODEL * DMODEL;
  bf16_t* wot  = wt + 3 * (size_t)DMODEL * DMODEL;
  bf16_t* qb   = (bf16_t*)(ws + 16 * MB);  // 8 MB  Q [B,H,L,64]
  bf16_t* kb   = (bf16_t*)(ws + 24 * MB);  // 8 MB  K [B,H,L,64]
  bf16_t* vtb  = (bf16_t*)(ws + 32 * MB);  // 8 MB  V^T [B,H,64,L]
  bf16_t* ctx  = (bf16_t*)(ws + 40 * MB);  // 8 MB  ctx [4096,1024]
  float*  bias = (float*)(ws + 48 * MB);   // 128 KB bias_delta [16,2048]

  k_convert_x<<<dim3((MROWS * DMODEL) / (256 * 4)), dim3(256), 0, stream>>>(hs, xb);
  k_transpose_w<<<dim3(32, 32, 4), dim3(32, 8), 0, stream>>>(wq, wk, wv, wo, wt);
  k_build_bias<<<dim3((HEADS * 2048) / 256), dim3(256), 0, stream>>>(rpb, bias);

  k_gemm_qkv<<<dim3(MROWS / 512, DMODEL / 64, 3), dim3(256), 0, stream>>>(
      xb, wqt, wkt, wvt, qb, kb, vtb);

  k_attention<<<dim3(SEQ / 64, BATCH * HEADS), dim3(128), 0, stream>>>(
      qb, kb, vtb, bias, ctx);

  k_gemm_out<<<dim3(MROWS / 512, DMODEL / 64), dim3(256), 0, stream>>>(ctx, wot, out);
}
